// TransformerBlock_41910290874805
// MI455X (gfx1250) — matrix operand order repaired and checked by reading, not compiled, not run on hardware
//
#include <hip/hip_runtime.h>

// ---------------------------------------------------------------------------
// CDNA5 (gfx1250) transformer block:
//   - all matmuls on v_wmma_f32_16x16x32_f16 (fp32 accumulate)
//   - GEMM A-tiles staged via TDM tensor_load_to_lds (TENSORcnt path) with
//     hardware LDS padding; per-lane global_load_async_to_lds_b128 fallback
//   - attention K-blocks staged via global_load_async_to_lds_b128 (ASYNCcnt)
//   - next-K-tile prefetch via global_prefetch_b8
// ---------------------------------------------------------------------------

typedef __attribute__((ext_vector_type(16))) _Float16     v16h;
typedef __attribute__((ext_vector_type(8)))  _Float16     v8h;
typedef __attribute__((ext_vector_type(8)))  float        v8f;
typedef __attribute__((ext_vector_type(4)))  unsigned int u32x4;
typedef __attribute__((ext_vector_type(8)))  int          i32x8;
typedef __attribute__((ext_vector_type(4)))  int          i32x4;

union V16H { v16h v; v8h h[2]; };

static __device__ __forceinline__ v8f wmma_f16(const V16H& a, const V16H& b, v8f c) {
  // D = A(16x32 f16) * B(32x16 f16) + C(16x16 f32)
  return __builtin_amdgcn_wmma_f32_16x16x32_f16(false, a.v, false, b.v,
                                                (short)0, c, false, false);
}

// Async memory->LDS 16-byte copy per lane (GLOBAL_LOAD_ASYNC_TO_LDS_B128,
// tracked with ASYNCcnt). LDS destination address = low 32 bits of the generic
// pointer to __shared__ (flat LDS aperture keeps the LDS offset in addr[31:0]).
static __device__ __forceinline__ void async_copy_b128(void* lds_dst,
                                                       const void* gsrc) {
  const unsigned int lds_off = (unsigned int)(unsigned long long)lds_dst;
  asm volatile("global_load_async_to_lds_b128 %0, %1, off"
               :: "v"(lds_off), "v"(gsrc)
               : "memory");
}

static __device__ __forceinline__ void wait_async_zero() {
#if __has_builtin(__builtin_amdgcn_s_wait_asynccnt)
  __builtin_amdgcn_s_wait_asynccnt(0);
#else
  asm volatile("s_wait_asynccnt 0x0" ::: "memory");
#endif
}

static __device__ __forceinline__ void wait_tensor_zero() {
#if __has_builtin(__builtin_amdgcn_s_wait_tensorcnt)
  __builtin_amdgcn_s_wait_tensorcnt(0);
#else
  asm volatile("s_wait_tensorcnt 0x0" ::: "memory");
#endif
}

#if __has_builtin(__builtin_amdgcn_tensor_load_to_lds)
#define HAVE_TDM 1
// Issue a 2D TDM tile load: tile_d1 x tile_d0 elements (2 bytes each) from a
// row-major tensor with row stride stride0 (elements) into LDS at lds_addr,
// with hardware padding pad_amount_code (DWORDs, encoded) inserted every
// 2^(pad_interval_code+1) DWORDs.  Descriptor per CDNA5 ISA sec 8.3/8.4.
// This toolchain exposes the 6-arg builtin:
//   (u32x4 g0, i32x8 g1, i32x4 g2, i32x4 g3, i32x8 g4, i32 cpol)
static __device__ __forceinline__ void tdm_load_2d(
    unsigned int lds_addr, unsigned long long gaddr,
    unsigned int tensor_d0, unsigned int tensor_d1,
    unsigned int tile_d0, unsigned int tile_d1,
    unsigned int stride0, unsigned int pad_interval_code,
    unsigned int pad_amount_code) {
  u32x4 g0;
  g0[0] = 1u;                                    // count=1, user mode
  g0[1] = lds_addr;                              // LDS byte address
  g0[2] = (unsigned int)(gaddr & 0xFFFFFFFFull); // global_addr[31:0]
  g0[3] = (unsigned int)((gaddr >> 32) & 0x01FFFFFFull) | (2u << 30); // type=2
  i32x8 g1;
  g1[0] = (int)((1u << 16)                       // data_size = 2 bytes
              | (1u << 20)                       // pad_enable
              | (pad_interval_code << 22)
              | (pad_amount_code << 25));
  g1[1] = (int)((tensor_d0 & 0xFFFFu) << 16);
  g1[2] = (int)((tensor_d0 >> 16) | ((tensor_d1 & 0xFFFFu) << 16));
  g1[3] = (int)((tensor_d1 >> 16) | (tile_d0 << 16));
  g1[4] = (int)tile_d1;                          // tile_dim1; tile_dim2=0
  g1[5] = (int)stride0;                          // tensor_dim0_stride[31:0]
  g1[6] = 0;
  g1[7] = 0;
  i32x4 z4; z4[0] = 0; z4[1] = 0; z4[2] = 0; z4[3] = 0;
  i32x8 z8;
#pragma unroll
  for (int i = 0; i < 8; ++i) z8[i] = 0;
  __builtin_amdgcn_tensor_load_to_lds(g0, g1, z4, z4, z8, 0);
}
#else
#define HAVE_TDM 0
#endif

constexpr int kD  = 1024;   // embed dim
constexpr int kH  = 16;     // heads
constexpr int kHD = 64;     // head dim
constexpr int kS  = 2048;   // sequence
constexpr int kB  = 2;      // batch
constexpr int kM  = kB * kS;  // 4096 rows
constexpr int kFF = 4096;   // ffn hidden

// ---------------------------------------------------------------------------
// fp32 -> f16 elementwise convert (weights)
// ---------------------------------------------------------------------------
__global__ void cvt_f32_to_f16(const float* __restrict__ in,
                               _Float16* __restrict__ out, int n) {
  const int i = blockIdx.x * 256 + threadIdx.x;
  if (i < n) out[i] = (_Float16)in[i];
}

// ---------------------------------------------------------------------------
// LayerNorm (unbiased variance, ddof=1, matching torch) -> f16 output
// ---------------------------------------------------------------------------
__global__ __launch_bounds__(256) void ln_kernel(
    const float* __restrict__ x, const float* __restrict__ sc,
    const float* __restrict__ sh, _Float16* __restrict__ out, int Ddim) {
  __shared__ float red[256];
  __shared__ float stats[2];
  const int row = blockIdx.x;
  const int tid = threadIdx.x;
  const float* xp = x + (size_t)row * Ddim;

  float s = 0.0f, ss = 0.0f;
  for (int j = tid; j < Ddim; j += 256) { const float v = xp[j]; s += v; ss += v * v; }

  red[tid] = s; __syncthreads();
  for (int o2 = 128; o2 > 0; o2 >>= 1) { if (tid < o2) red[tid] += red[tid + o2]; __syncthreads(); }
  if (tid == 0) stats[0] = red[0];
  __syncthreads();
  red[tid] = ss; __syncthreads();
  for (int o2 = 128; o2 > 0; o2 >>= 1) { if (tid < o2) red[tid] += red[tid + o2]; __syncthreads(); }
  if (tid == 0) stats[1] = red[0];
  __syncthreads();

  const float mean = stats[0] / (float)Ddim;
  const float var  = (stats[1] - (float)Ddim * mean * mean) / (float)(Ddim - 1);
  const float rstd = rsqrtf(var + 1e-5f);
  _Float16* op = out + (size_t)row * Ddim;
  for (int j = tid; j < Ddim; j += 256)
    op[j] = (_Float16)((xp[j] - mean) * rstd * sc[j] + sh[j]);
}

// ---------------------------------------------------------------------------
// WMMA GEMM: C[M,N] = A[M,K](f16) * B[K,N](f16)  (+bias)(+gelu)(+fp32 residual)
// block = 256 thr (8 waves); block tile 128x128; wave tile 32x64 (2x4 wmma).
// A tile staged by TDM (with HW LDS padding) or per-lane async copies.
// B tile transposed on ds-store.  Fragment layouts (A/B use contiguous k, not the ISA table order):
//   A (16x32): lane r=lane&15 -> row; elem e: K = hi*16 + e
//   B (32x16): lane r -> column; elem e: K = hi*16 + e   (B staged transposed)
//   D (16x16): lane = n + 16*(m>=8); vgpr v -> m = hi*8 + v
// ---------------------------------------------------------------------------
template<int BIAS, int RES, int DO_GELU, int STORE_F32, int STORE_F16>
__global__ __launch_bounds__(256) void gemm_wmma_kernel(
    const _Float16* __restrict__ A, const _Float16* __restrict__ Bw,
    const float* __restrict__ bias, const float* __restrict__ res,
    float* __restrict__ outF, _Float16* __restrict__ outH,
    int Mdim, int Ndim, int Kdim) {
  __shared__ _Float16 As[128 * 40];   // [row][k] stride 40 halves (pad)
  __shared__ _Float16 Bt[128 * 40];   // [n][k]   transposed, stride 40

  const int tid  = threadIdx.x;
  const int wid  = tid >> 5;
  const int lane = tid & 31;
  const int r    = lane & 15;
  const int hi   = lane >> 4;
  const int wm   = wid >> 1;        // 0..3 -> M (4 x 32 rows)
  const int wn   = wid & 1;         // 0..1 -> N (2 x 64 cols)
  const int bm   = blockIdx.y * 128;
  const int bn   = blockIdx.x * 128;

  // per-thread staging coordinates
  const int arow = tid >> 1;              // 0..127
  const int acol = (tid & 1) * 16;        // 0 / 16
  const int bk   = tid >> 3;              // 0..31
  const int bnc  = (tid & 7) * 16;        // 0..112

  v8f acc[2][4];
#pragma unroll
  for (int mi = 0; mi < 2; ++mi)
#pragma unroll
    for (int ni = 0; ni < 4; ++ni)
#pragma unroll
      for (int e = 0; e < 8; ++e) acc[mi][ni][e] = 0.0f;

  for (int k0 = 0; k0 < Kdim; k0 += 32) {
    __syncthreads();
#if HAVE_TDM
    if (wid == 0) { // one wave issues the tile DMA; TDM ignores EXEC
      const unsigned long long ga =
          (unsigned long long)(const void*)(A + (size_t)bm * Kdim + k0);
      tdm_load_2d((unsigned int)(unsigned long long)(void*)As, ga,
                  (unsigned int)Kdim, (unsigned int)Mdim,
                  /*tile_d0=*/32u, /*tile_d1=*/128u,
                  /*stride0=*/(unsigned int)Kdim,
                  /*pad every 16 dwords=*/3u, /*pad 4 dwords=*/3u);
    }
#else
    { // stage A tile 128x32: async memory->LDS, 2 x 16B per thread
      const _Float16* ap = A + (size_t)(bm + arow) * Kdim + k0 + acol;
      async_copy_b128(&As[arow * 40 + acol],     ap);
      async_copy_b128(&As[arow * 40 + acol + 8], ap + 8);
      if (k0 + 32 < Kdim) __builtin_prefetch(ap + 32, 0, 3);
    }
#endif
    { // stage B tile 32x128 transposed -> Bt[n][k]
      const _Float16* bp = Bw + (size_t)(k0 + bk) * Ndim + bn + bnc;
      v8h b0 = *(const v8h*)bp;
      v8h b1 = *(const v8h*)(bp + 8);
#pragma unroll
      for (int i = 0; i < 8; ++i) {
        Bt[(bnc + i) * 40 + bk]     = b0[i];
        Bt[(bnc + 8 + i) * 40 + bk] = b1[i];
      }
      if (k0 + 32 < Kdim) __builtin_prefetch(bp + (size_t)32 * Ndim, 0, 3);
    }
#if HAVE_TDM
    if (wid == 0) wait_tensor_zero();
#else
    wait_async_zero();
#endif
    __syncthreads();

    V16H af[2], bf[4];
#pragma unroll
    for (int mi = 0; mi < 2; ++mi) {
      const _Float16* p = &As[(wm * 32 + mi * 16 + r) * 40];
      af[mi].h[0] = *(const v8h*)(p + hi * 16);
      af[mi].h[1] = *(const v8h*)(p + hi * 16 + 8);
    }
#pragma unroll
    for (int ni = 0; ni < 4; ++ni) {
      const _Float16* p = &Bt[(wn * 64 + ni * 16 + r) * 40 + hi * 16];
      bf[ni].h[0] = *(const v8h*)p;
      bf[ni].h[1] = *(const v8h*)(p + 8);
    }
#pragma unroll
    for (int mi = 0; mi < 2; ++mi)
#pragma unroll
      for (int ni = 0; ni < 4; ++ni)
        acc[mi][ni] = wmma_f16(af[mi], bf[ni], acc[mi][ni]);
  }

#pragma unroll
  for (int mi = 0; mi < 2; ++mi)
#pragma unroll
    for (int ni = 0; ni < 4; ++ni)
#pragma unroll
      for (int v = 0; v < 8; ++v) {
        const int m = bm + wm * 32 + mi * 16 + hi * 8 + v;
        const int n = bn + wn * 64 + ni * 16 + r;
        float val = acc[mi][ni][v];
        if (BIAS) val += bias[n];
        if (DO_GELU) {
          const float t = val;
          val = 0.5f * t *
                (1.0f + tanhf(0.7978845608028654f * (t + 0.044715f * t * t * t)));
        }
        if (RES) val += res[(size_t)m * Ndim + n];
        if (STORE_F32) outF[(size_t)m * Ndim + n] = val;
        if (STORE_F16) outH[(size_t)m * Ndim + n] = (_Float16)val;
      }
}

// ---------------------------------------------------------------------------
// Flash-style causal attention. Grid: (S/128, H, B); 256 thr = 8 waves, each
// wave owns 16 q-rows. KV processed in 64-key blocks: K block staged into LDS
// with async copies (no transpose needed), V block transposed on ds-store.
// Mask applied BEFORE the 1/sqrt(hd) scale.  fp32 online softmax.
// ---------------------------------------------------------------------------
__global__ __launch_bounds__(256) void attn_kernel(
    const _Float16* __restrict__ Q, const _Float16* __restrict__ K,
    const _Float16* __restrict__ V, _Float16* __restrict__ ctx) {
  __shared__ _Float16 Kls[64 * 72];       // K block row-major: [key][d]
  __shared__ _Float16 Vt[64 * 72];        // V block transposed: [d][key]
  __shared__ _Float16 Pls[8 * 16 * 72];   // per-wave P tile: [qrow][key]

  const int b   = blockIdx.z;
  const int h   = blockIdx.y;
  const int q0  = blockIdx.x * 128;
  const int tid = threadIdx.x;
  const int w   = tid >> 5;
  const int lane = tid & 31;
  const int r   = lane & 15;
  const int hi  = lane >> 4;

  const size_t headBase = (size_t)b * kS * kD + (size_t)h * kHD;

  // Q fragments for this wave's 16 rows (hd=64 -> two K-steps of 32)
  V16H aq[2];
  {
    const _Float16* qp = Q + headBase + (size_t)(q0 + w * 16 + r) * kD;
#pragma unroll
    for (int j = 0; j < 2; ++j) {
      aq[j].h[0] = *(const v8h*)(qp + j * 32 + hi * 16);
      aq[j].h[1] = *(const v8h*)(qp + j * 32 + hi * 16 + 8);
    }
  }

  float mrun[8], lrun[8];
  v8f o[4];
#pragma unroll
  for (int v = 0; v < 8; ++v) { mrun[v] = -__builtin_inff(); lrun[v] = 0.0f; }
#pragma unroll
  for (int nt = 0; nt < 4; ++nt)
#pragma unroll
    for (int e = 0; e < 8; ++e) o[nt][e] = 0.0f;

  const int wave_qmax = q0 + w * 16 + 15;
  const int kv_end = q0 + 128;   // causal upper bound for this block

  for (int kb = 0; kb < kv_end; kb += 64) {
    __syncthreads();
    {
      const int krow = tid & 63;
      const int dc   = (tid >> 6) * 16;
      // ---- K block: async memory->LDS, row-major (no transpose needed) ----
      const _Float16* kpg = K + headBase + (size_t)(kb + krow) * kD + dc;
      async_copy_b128(&Kls[krow * 72 + dc],     kpg);
      async_copy_b128(&Kls[krow * 72 + dc + 8], kpg + 8);
      // ---- V block: transpose on ds-store ----
      const _Float16* vp = V + headBase + (size_t)(kb + krow) * kD + dc;
      v8h v0 = *(const v8h*)vp;
      v8h v1 = *(const v8h*)(vp + 8);
#pragma unroll
      for (int i = 0; i < 8; ++i) {
        Vt[(dc + i) * 72 + krow]     = v0[i];
        Vt[(dc + 8 + i) * 72 + krow] = v1[i];
      }
    }
    wait_async_zero();
    __syncthreads();

    if (kb <= wave_qmax) {   // wave-uniform: EXEC stays all-ones for WMMA
      // ---- scores S = Q K^T (16 x 64) ----
      v8f s[4];
#pragma unroll
      for (int ni = 0; ni < 4; ++ni) {
#pragma unroll
        for (int e = 0; e < 8; ++e) s[ni][e] = 0.0f;
#pragma unroll
        for (int j = 0; j < 2; ++j) {
          V16H bk;  // B frag: lane -> key column; K rows contiguous over hd
          const _Float16* kp = &Kls[(ni * 16 + r) * 72 + j * 32 + hi * 16];
          bk.h[0] = *(const v8h*)kp;
          bk.h[1] = *(const v8h*)(kp + 8);
          s[ni] = wmma_f16(aq[j], bk, s[ni]);
        }
      }

      // ---- mask, scale, online softmax (per D-tile row m = hi*8+v) ----
      float alpha[8];
#pragma unroll
      for (int v = 0; v < 8; ++v) {
        const int qr = q0 + w * 16 + hi * 8 + v;
        float xs[4];
        float rm = -__builtin_inff();
#pragma unroll
        for (int ni = 0; ni < 4; ++ni) {
          const int key = kb + ni * 16 + r;
          float sv = s[ni][v];
          sv = (key > qr) ? -__builtin_inff() : sv * 0.125f;  // mask, then /sqrt(64)
          xs[ni] = sv;
          rm = fmaxf(rm, sv);
        }
#pragma unroll
        for (int off = 1; off <= 8; off <<= 1)
          rm = fmaxf(rm, __shfl_xor(rm, off, 32));
        const float mnew = fmaxf(mrun[v], rm);
        const float al = expf(mrun[v] - mnew);
        float rs = 0.0f;
#pragma unroll
        for (int ni = 0; ni < 4; ++ni) {
          const float p = expf(xs[ni] - mnew);
          rs += p;
          Pls[(w * 16 + hi * 8 + v) * 72 + ni * 16 + r] = (_Float16)p;
        }
#pragma unroll
        for (int off = 1; off <= 8; off <<= 1)
          rs += __shfl_xor(rs, off, 32);
        lrun[v] = lrun[v] * al + rs;
        mrun[v] = mnew;
        alpha[v] = al;
      }

#pragma unroll
      for (int nt = 0; nt < 4; ++nt)
#pragma unroll
        for (int v = 0; v < 8; ++v) o[nt][v] *= alpha[v];

      // ---- O += P @ V (16 x 64 keys -> 16 x 64 hd) ----
#pragma unroll
      for (int j2 = 0; j2 < 2; ++j2) {
        V16H ap;   // A frag of P from wave-private LDS (in-order DS per wave)
        const _Float16* pp = &Pls[(w * 16 + r) * 72 + j2 * 32];
        ap.h[0] = *(const v8h*)(pp + hi * 16);
        ap.h[1] = *(const v8h*)(pp + hi * 16 + 8);
#pragma unroll
        for (int nt = 0; nt < 4; ++nt) {
          V16H bv_;  // B frag of V from transposed LDS tile
          const _Float16* vp2 = &Vt[(nt * 16 + r) * 72 + j2 * 32 + hi * 16];
          bv_.h[0] = *(const v8h*)vp2;
          bv_.h[1] = *(const v8h*)(vp2 + 8);
          o[nt] = wmma_f16(ap, bv_, o[nt]);
        }
      }
    }
  }

  // ---- normalize and store ctx (f16, [B,S,D] with head offset) ----
#pragma unroll
  for (int nt = 0; nt < 4; ++nt)
#pragma unroll
    for (int v = 0; v < 8; ++v) {
      const int qr = q0 + w * 16 + hi * 8 + v;
      const float val = o[nt][v] / lrun[v];
      ctx[headBase + (size_t)qr * kD + nt * 16 + r] = (_Float16)val;
    }
}

// ---------------------------------------------------------------------------
// Host launch
// ---------------------------------------------------------------------------
extern "C" void kernel_launch(void* const* d_in, const int* in_sizes, int n_in,
                              void* d_out, int out_size, void* d_ws, size_t ws_size,
                              hipStream_t stream) {
  (void)in_sizes; (void)n_in; (void)out_size; (void)ws_size;

  const float* x    = (const float*)d_in[0];
  const float* ln1s = (const float*)d_in[1];
  const float* ln1b = (const float*)d_in[2];
  const float* wq   = (const float*)d_in[3];
  const float* wk   = (const float*)d_in[4];
  const float* wv   = (const float*)d_in[5];
  const float* wo   = (const float*)d_in[6];
  const float* bo   = (const float*)d_in[7];
  const float* ln2s = (const float*)d_in[8];
  const float* ln2b = (const float*)d_in[9];
  const float* w1   = (const float*)d_in[10];
  const float* b1   = (const float*)d_in[11];
  const float* w2   = (const float*)d_in[12];
  const float* b2   = (const float*)d_in[13];

  char* base = (char*)d_ws;
  size_t off = 0;
  auto carve = [&](size_t bytes) -> char* {
    char* q = base + off;
    off += (bytes + 255) & ~(size_t)255;
    return q;
  };

  _Float16* wq_h  = (_Float16*)carve((size_t)kD * kD * 2);
  _Float16* wk_h  = (_Float16*)carve((size_t)kD * kD * 2);
  _Float16* wv_h  = (_Float16*)carve((size_t)kD * kD * 2);
  _Float16* wo_h  = (_Float16*)carve((size_t)kD * kD * 2);
  _Float16* w1_h  = (_Float16*)carve((size_t)kD * kFF * 2);
  _Float16* w2_h  = (_Float16*)carve((size_t)kFF * kD * 2);
  _Float16* ln1_h = (_Float16*)carve((size_t)kM * kD * 2);
  _Float16* q_h   = (_Float16*)carve((size_t)kM * kD * 2);
  _Float16* k_h   = (_Float16*)carve((size_t)kM * kD * 2);
  _Float16* v_h   = (_Float16*)carve((size_t)kM * kD * 2);
  _Float16* ctx_h = (_Float16*)carve((size_t)kM * kD * 2);
  _Float16* ln2_h = (_Float16*)carve((size_t)kM * kD * 2);
  _Float16* h1_h  = (_Float16*)carve((size_t)kM * kFF * 2);
  float*    resid = (float*)   carve((size_t)kM * kD * 4);

  // 1) weight conversions fp32 -> f16
  const int wdd = kD * kD, wdf = kD * kFF;
  cvt_f32_to_f16<<<(wdd + 255) / 256, 256, 0, stream>>>(wq, wq_h, wdd);
  cvt_f32_to_f16<<<(wdd + 255) / 256, 256, 0, stream>>>(wk, wk_h, wdd);
  cvt_f32_to_f16<<<(wdd + 255) / 256, 256, 0, stream>>>(wv, wv_h, wdd);
  cvt_f32_to_f16<<<(wdd + 255) / 256, 256, 0, stream>>>(wo, wo_h, wdd);
  cvt_f32_to_f16<<<(wdf + 255) / 256, 256, 0, stream>>>(w1, w1_h, wdf);
  cvt_f32_to_f16<<<(wdf + 255) / 256, 256, 0, stream>>>(w2, w2_h, wdf);

  // 2) LN1
  ln_kernel<<<kM, 256, 0, stream>>>(x, ln1s, ln1b, ln1_h, kD);

  // 3) QKV projections (no bias, store f16)
  dim3 gP(kD / 128, kM / 128);
  gemm_wmma_kernel<0,0,0,0,1><<<gP, 256, 0, stream>>>(ln1_h, wq_h, nullptr, nullptr,
                                                      nullptr, q_h, kM, kD, kD);
  gemm_wmma_kernel<0,0,0,0,1><<<gP, 256, 0, stream>>>(ln1_h, wk_h, nullptr, nullptr,
                                                      nullptr, k_h, kM, kD, kD);
  gemm_wmma_kernel<0,0,0,0,1><<<gP, 256, 0, stream>>>(ln1_h, wv_h, nullptr, nullptr,
                                                      nullptr, v_h, kM, kD, kD);

  // 4) causal attention
  attn_kernel<<<dim3(kS / 128, kH, kB), 256, 0, stream>>>(q_h, k_h, v_h, ctx_h);

  // 5) out-proj + b_out + residual(x) -> resid (fp32)
  gemm_wmma_kernel<1,1,0,1,0><<<gP, 256, 0, stream>>>(ctx_h, wo_h, bo, x,
                                                      resid, nullptr, kM, kD, kD);

  // 6) LN2
  ln_kernel<<<kM, 256, 0, stream>>>(resid, ln2s, ln2b, ln2_h, kD);

  // 7) FFN1 + b1 + GELU(tanh) -> h1 (f16)
  gemm_wmma_kernel<1,0,1,0,1><<<dim3(kFF / 128, kM / 128), 256, 0, stream>>>(
      ln2_h, w1_h, b1, nullptr, nullptr, h1_h, kM, kFF, kD);

  // 8) FFN2 + b2 + residual(resid) -> d_out (fp32)
  gemm_wmma_kernel<1,1,0,1,0><<<gP, 256, 0, stream>>>(
      h1_h, w2_h, b2, resid, (float*)d_out, nullptr, kM, kD, kFF);
}
